// ElParoFlowNetCvEa_69904887710364
// MI455X (gfx1250) — matrix operand order repaired and checked by reading, not compiled, not run on hardware
//
#include <hip/hip_runtime.h>
#include <stddef.h>

typedef __bf16 bf16;
typedef __bf16 v16bf __attribute__((ext_vector_type(16)));
typedef __bf16 v8bf  __attribute__((ext_vector_type(8)));
typedef float  v8f   __attribute__((ext_vector_type(8)));

#define BN_EPS 1e-5f

// lrelu(v) = max(v, 0.01v): single v_max_num_f32 instead of cmp+cndmask
__device__ __forceinline__ float lrelu(float v) { return fmaxf(v, 0.01f * v); }

// ---------------------------------------------------------------------------
// conv1: 7x7, 1->32, pad 2, BN + leaky-relu, fp32 in -> bf16 [P][32] out
// ---------------------------------------------------------------------------
__global__ void k_conv7(const float* __restrict__ in,   // [8][1][128][128]
                        const float* __restrict__ w,    // [32][1][7][7]
                        const float* __restrict__ b,    // [32]
                        const float* __restrict__ g,    // [32]
                        const float* __restrict__ bt,   // [32]
                        bf16* __restrict__ out)         // [8*126*126][32]
{
    int p = blockIdx.x * blockDim.x + threadIdx.x;
    const int P = 8 * 126 * 126;
    if (p >= P) return;
    int x = p % 126;
    int y = (p / 126) % 126;
    int bn = p / (126 * 126);
    const float* img = in + (size_t)bn * 128 * 128;

    float acc[32];
#pragma unroll
    for (int c = 0; c < 32; ++c) acc[c] = 0.f;

    for (int ky = 0; ky < 7; ++ky) {
        int iy = y + ky - 2;
        if (iy < 0 || iy >= 128) continue;
        for (int kx = 0; kx < 7; ++kx) {
            int ix = x + kx - 2;
            if (ix < 0 || ix >= 128) continue;
            float v = img[iy * 128 + ix];
#pragma unroll
            for (int c = 0; c < 32; ++c)
                acc[c] += v * w[c * 49 + ky * 7 + kx];
        }
    }

    bf16* o = out + (size_t)p * 32;
    float rs = rsqrtf(1.f + BN_EPS);
#pragma unroll
    for (int c = 0; c < 32; ++c) {
        float s = g[c] * rs;
        o[c] = (bf16)lrelu(acc[c] * s + (b[c] * s + bt[c]));
    }
}

// ---------------------------------------------------------------------------
// Pack conv weights (fp32, BN-scale folded) into WMMA A-fragment layout:
//   wp[((tap*KT + kt)*Mt + mt)*512 + lane*16 + e]
//   lane l, elem e  ->  M = mt*16 + l%16,
//                       K = kt*32 + 16*(l/16) + e
// Raw weight flat index: (o*Ci + k)*T + tap  (covers OIHW 1x1, OIDHW, 3x3)
// ---------------------------------------------------------------------------
__global__ void k_pack_w(const float* __restrict__ w, const float* __restrict__ g,
                         bf16* __restrict__ wp,
                         int Ci, int Co, int KT, int Mt, int T)
{
    long i = (long)blockIdx.x * blockDim.x + threadIdx.x;
    long total = (long)T * KT * Mt * 512;
    if (i >= total) return;
    int e = (int)(i & 15);
    int lane = (int)((i >> 4) & 31);
    long rest = i >> 9;
    int mt = (int)(rest % Mt); rest /= Mt;
    int kt = (int)(rest % KT);
    int tap = (int)(rest / KT);
    int o = mt * 16 + (lane & 15);
    int k = kt * 32 + ((lane >> 4) * 16) + e;
    float val = 0.f;
    if (o < Co && k < Ci) {
        float s = g[o] * rsqrtf(1.f + BN_EPS);
        val = w[(long)(o * Ci + k) * T + tap] * s;
    }
    wp[i] = (bf16)val;
}

__global__ void k_pack_bias(const float* __restrict__ b, const float* __restrict__ g,
                            const float* __restrict__ bt, float* __restrict__ bp,
                            int Co, int Cpad)
{
    int o = blockIdx.x * blockDim.x + threadIdx.x;
    if (o >= Cpad) return;
    float v = 0.f;
    if (o < Co) {
        float s = g[o] * rsqrtf(1.f + BN_EPS);
        v = b[o] * s + bt[o];
    }
    bp[o] = v;
}

__global__ void k_zero_bf16(bf16* __restrict__ z, int n)
{
    int i = blockIdx.x * blockDim.x + threadIdx.x;
    if (i < n) z[i] = (bf16)0.f;
}

// ---------------------------------------------------------------------------
// Cost volume: mv[p][dyi*17+dxi] = mask ? 1/sum_c|fa-fb_shift| : 0, bf16.
// One thread per (pixel, dyi). Also zero-fills padded channels 289..319.
// ---------------------------------------------------------------------------
__global__ void k_cost(const bf16* __restrict__ fa, const bf16* __restrict__ fb,
                       bf16* __restrict__ mv)
{
    long t = (long)blockIdx.x * blockDim.x + threadIdx.x;
    const int P = 8 * 126 * 126;
    if (t >= (long)P * 17) return;
    int dyi = (int)(t % 17);
    long p = t / 17;
    int x = (int)(p % 126);
    int y = (int)((p / 126) % 126);
    int bn = (int)(p / (126 * 126));
    int dy = dyi - 8;

    float a[32];
    const bf16* ap = fa + p * 32;
#pragma unroll
    for (int c = 0; c < 32; ++c) a[c] = (float)ap[c];

    bf16* o = mv + p * 320 + dyi * 17;
    int sy = y - dy;
    bool vy = (sy >= 0) && (sy < 126);
    for (int dxi = 0; dxi < 17; ++dxi) {
        int sx = x - (dxi - 8);
        float r = 0.f;
        if (vy && sx >= 0 && sx < 126) {
            const bf16* bp = fb + (long)((bn * 126 + sy) * 126 + sx) * 32;
            float cst = 0.f;
#pragma unroll
            for (int c = 0; c < 32; ++c) cst += fabsf(a[c] - (float)bp[c]);
            r = 1.f / cst;
        }
        o[dxi] = (bf16)r;
    }
    if (dyi == 16) {
        bf16* tail = mv + p * 320 + 289;
        for (int k = 0; k < 31; ++k) tail[k] = (bf16)0.f;
    }
}

// ---------------------------------------------------------------------------
// Implicit-GEMM WMMA conv (1x1 / 3x3), fused bias + leaky-relu.
//   X: [Pin][CinPad] bf16 (channel-minor), Y: [Pout][CoutPad] bf16
//   Wp: packed A fragments, Bp: folded bias [CoutPad], zbuf: >=CinPad zeros
// Fully specialized on <TAPS, KT, MB>: unconditional B loads for 1x1,
// zero-page pointer select for 3x3 halo (EXEC stays all-ones), one-step
// software pipeline on the B fragment, MB A-tiles per B load.
// ---------------------------------------------------------------------------
template <int TAPS, int KT, int MB>
__global__ void k_gemm(const bf16* __restrict__ X, bf16* __restrict__ Y,
                       const bf16* __restrict__ Wp, const float* __restrict__ Bp,
                       const bf16* __restrict__ zbuf,
                       int Mt, int CinPad, int CoutPad, int pad,
                       int H, int W, int OH, int OW, int Pout)
{
    extern __shared__ bf16 smem[];
    const int tid = threadIdx.x;
    const int lane = tid & 31;
    const int wid = tid >> 5;

    int Ptiles = Pout >> 4;
    int PG = (Ptiles + 7) >> 3;
    int mblock = blockIdx.x / PG;
    int pgroup = blockIdx.x % PG;

    // Stage this M-block's A fragments into LDS.
    {
        constexpr int TOTAL = TAPS * KT * MB * 512;
        for (int i = tid; i < TOTAL; i += 256) {
            int j = i & 511;
            int rest = i >> 9;
            int mtl = rest % MB; rest /= MB;
            int kt = rest % KT;
            int tap = rest / KT;
            int mt = mblock * MB + mtl;
            smem[i] = Wp[((long)(tap * KT + kt) * Mt + mt) * 512 + j];
        }
    }
    __syncthreads();

    int ptile = pgroup * 8 + wid;
    if (ptile >= Ptiles) return;

    int p = ptile * 16 + (lane & 15);           // this lane's output pixel
    int ox = p % OW;
    int oy = (p / OW) % OH;
    int ob = p / (OW * OH);
    int khalf = (lane >> 4) * 16;               // B-fragment K offset per half

    v8f acc[MB];
#pragma unroll
    for (int m = 0; m < MB; ++m)
#pragma unroll
        for (int r = 0; r < 8; ++r) acc[m][r] = 0.f;

#pragma unroll
    for (int tap = 0; tap < TAPS; ++tap) {
        const bf16* base;
        if (TAPS == 1) {
            base = X + (size_t)p * CinPad + khalf;
        } else {
            int iy = oy + tap / 3 - pad;
            int ix = ox + tap % 3 - pad;
            bool valid = (iy >= 0) && (iy < H) && (ix >= 0) && (ix < W);
            const bf16* xp = X + (size_t)((ob * H + iy) * W + ix) * CinPad + khalf;
            base = valid ? xp : (zbuf + khalf);   // zero page for halo lanes
        }
        const bf16* sA = smem + (size_t)tap * KT * MB * 512 + lane * 16;

        v16bf bcur = *(const v16bf*)(base);
#pragma unroll
        for (int kt = 0; kt < KT; ++kt) {
            v16bf bfrag = bcur;
            if (kt + 1 < KT)
                bcur = *(const v16bf*)(base + (kt + 1) * 32);
#pragma unroll
            for (int m = 0; m < MB; ++m) {
                v16bf afrag = *(const v16bf*)(sA + (size_t)(kt * MB + m) * 512);
                acc[m] = __builtin_amdgcn_wmma_f32_16x16x32_bf16(
                    false, afrag, false, bfrag, (short)0, acc[m], false, false);
            }
        }
    }

    // Epilogue: bias + lrelu (v_max), f32 -> bf16, contiguous 16B stores.
#pragma unroll
    for (int m = 0; m < MB; ++m) {
        int coBase = (mblock * MB + m) * 16 + (lane >> 4) * 8;
        v8bf ov;
#pragma unroll
        for (int r = 0; r < 8; ++r)
            ov[r] = (bf16)lrelu(acc[m][r] + Bp[coBase + r]);
        *(v8bf*)(Y + (size_t)p * CoutPad + coBase) = ov;
    }
}

// ---------------------------------------------------------------------------
// Final 18->2 conv (no BN/relu), bf16 [P2][32] -> fp32 NCHW (8,2,124,124)
// ---------------------------------------------------------------------------
__global__ void k_final(const bf16* __restrict__ X, const float* __restrict__ w,
                        const float* __restrict__ b, float* __restrict__ out)
{
    int p = blockIdx.x * blockDim.x + threadIdx.x;
    const int P = 8 * 124 * 124;
    if (p >= P) return;
    int x = p % 124;
    int y = (p / 124) % 124;
    int bn = p / (124 * 124);
    const bf16* xp = X + (size_t)p * 32;
    float a0 = b[0], a1 = b[1];
#pragma unroll
    for (int k = 0; k < 18; ++k) {
        float v = (float)xp[k];
        a0 += w[k] * v;
        a1 += w[18 + k] * v;
    }
    out[((size_t)(bn * 2 + 0) * 124 + y) * 124 + x] = a0;
    out[((size_t)(bn * 2 + 1) * 124 + y) * 124 + x] = a1;
}

// ---------------------------------------------------------------------------
static void launch_gemm(const bf16* X, bf16* Y, const bf16* Wp, const float* Bp,
                        const bf16* zbuf, int K, int Co, int taps, int MB, int pad,
                        int H, int W, int OH, int OW, hipStream_t stream)
{
    int KT = (K + 31) / 32;
    int Cpad = ((Co + 31) / 32) * 32;
    int Mt = Cpad / 16;
    int CinPad = KT * 32;
    int Pout = 8 * OH * OW;
    int Ptiles = Pout / 16;
    int PG = (Ptiles + 7) / 8;
    int MBlocks = Mt / MB;
    dim3 grid(MBlocks * PG), block(256);
    size_t smem = (size_t)taps * KT * MB * 512 * sizeof(bf16);

#define GEMM_CASE(T, KTC, M)                                                 \
    if (taps == T && KT == KTC && MB == M) {                                 \
        k_gemm<T, KTC, M><<<grid, block, smem, stream>>>(                    \
            X, Y, Wp, Bp, zbuf, Mt, CinPad, Cpad, pad, H, W, OH, OW, Pout); \
        return;                                                              \
    }
    GEMM_CASE(1, 1, 2)
    GEMM_CASE(1, 10, 4)
    GEMM_CASE(1, 10, 5)
    GEMM_CASE(1, 5, 3)
    GEMM_CASE(9, 3, 2)
    GEMM_CASE(1, 3, 4)
    GEMM_CASE(9, 2, 2)
    GEMM_CASE(1, 2, 2)
#undef GEMM_CASE
}

extern "C" void kernel_launch(void* const* d_in, const int* in_sizes, int n_in,
                              void* d_out, int out_size, void* d_ws, size_t ws_size,
                              hipStream_t stream)
{
    (void)in_sizes; (void)n_in; (void)out_size; (void)ws_size;
    const float* pa = (const float*)d_in[0];
    const float* pb = (const float*)d_in[1];
    // feat layer i params at d_in[2+4i..]; flow layer j at d_in[22+4j..]; final 62,63

    const int P1 = 8 * 126 * 126;
    const int P2 = 8 * 124 * 124;

    size_t off = 0;
    char* wsb = (char*)d_ws;
    auto carve = [&](size_t bytes) -> void* {
        void* p = wsb + off;
        off = (off + bytes + 255) & ~(size_t)255;
        return p;
    };

    bf16* fa0 = (bf16*)carve((size_t)P1 * 32 * 2);
    bf16* fa1 = (bf16*)carve((size_t)P1 * 32 * 2);
    bf16* fb0 = (bf16*)carve((size_t)P1 * 32 * 2);
    bf16* fb1 = (bf16*)carve((size_t)P1 * 32 * 2);
    bf16* M0  = (bf16*)carve((size_t)P1 * 320 * 2);
    bf16* M1  = (bf16*)carve((size_t)P1 * 320 * 2);
    bf16* zbuf = (bf16*)carve(512 * 2);   // zero page for 3x3 halo lanes

    k_zero_bf16<<<dim3(2), dim3(256), 0, stream>>>(zbuf, 512);

    // GEMM layers: 0..3 = feature 1x1 chain, 4..13 = flow net
    int gK[14]    = {32, 32, 24, 24, 289, 289, 144, 72, 144, 72, 36, 144, 72, 36};
    int gCo[14]   = {32, 24, 24, 16, 289, 144, 72, 144, 72, 36, 144, 72, 36, 18};
    int gTaps[14] = { 1,  1,  1,  1,   1,   1,  1,   9,  1,  1,   9,  1,  1,  1};
    int gMB[14]   = { 2,  2,  2,  2,   4,   5,  3,   2,  3,  4,   2,  3,  4,  2};
    int gW[14];
    for (int i = 0; i < 4; ++i)  gW[i] = 2 + 4 * (i + 1);
    for (int j = 0; j < 10; ++j) gW[4 + j] = 22 + 4 * j;

    bf16* Wp[14]; float* Bp[14];
    for (int l = 0; l < 14; ++l) {
        int KT = (gK[l] + 31) / 32;
        int Cpad = ((gCo[l] + 31) / 32) * 32;
        int Mt = Cpad / 16;
        size_t nw = (size_t)gTaps[l] * KT * Mt * 512;
        Wp[l] = (bf16*)carve(nw * 2);
        Bp[l] = (float*)carve((size_t)Cpad * 4);
        const float* w  = (const float*)d_in[gW[l]];
        const float* bb = (const float*)d_in[gW[l] + 1];
        const float* gg = (const float*)d_in[gW[l] + 2];
        const float* tt = (const float*)d_in[gW[l] + 3];
        k_pack_w<<<dim3((unsigned)((nw + 255) / 256)), dim3(256), 0, stream>>>(
            w, gg, Wp[l], gK[l], gCo[l], KT, Mt, gTaps[l]);
        k_pack_bias<<<dim3((Cpad + 255) / 256), dim3(256), 0, stream>>>(
            bb, gg, tt, Bp[l], gCo[l], Cpad);
    }

    // Feature net
    k_conv7<<<dim3((P1 + 255) / 256), dim3(256), 0, stream>>>(
        pa, (const float*)d_in[2], (const float*)d_in[3],
        (const float*)d_in[4], (const float*)d_in[5], fa0);
    k_conv7<<<dim3((P1 + 255) / 256), dim3(256), 0, stream>>>(
        pb, (const float*)d_in[2], (const float*)d_in[3],
        (const float*)d_in[4], (const float*)d_in[5], fb0);

    bf16 *xa = fa0, *ya = fa1, *xb = fb0, *yb = fb1;
    for (int l = 0; l < 4; ++l) {
        launch_gemm(xa, ya, Wp[l], Bp[l], zbuf, gK[l], gCo[l], 1, gMB[l], 0,
                    126, 126, 126, 126, stream);
        launch_gemm(xb, yb, Wp[l], Bp[l], zbuf, gK[l], gCo[l], 1, gMB[l], 0,
                    126, 126, 126, 126, stream);
        bf16* t;
        t = xa; xa = ya; ya = t;
        t = xb; xb = yb; yb = t;
    }
    // After 4 swaps xa == fa0, xb == fb0 hold final 16-ch features (padded to 32).

    // Cost volume -> M0 as [P1][320] bf16
    long costThreads = (long)P1 * 17;
    k_cost<<<dim3((unsigned)((costThreads + 255) / 256)), dim3(256), 0, stream>>>(xa, xb, M0);

    // Flow net
    bf16 *X = M0, *Y = M1;
    for (int j = 0; j < 10; ++j) {
        int l = 4 + j;
        int pad = (j == 3) ? 1 : 0;
        int H  = (j <= 6) ? 126 : 124;   // input grid
        int OS = (j <= 5) ? 126 : 124;   // output grid
        launch_gemm(X, Y, Wp[l], Bp[l], zbuf, gK[l], gCo[l], gTaps[l], gMB[l], pad,
                    H, H, OS, OS, stream);
        bf16* t = X; X = Y; Y = t;
    }
    // X now holds flow layer 9 output: [P2][32] (channels 18..31 zero)

    k_final<<<dim3((P2 + 255) / 256), dim3(256), 0, stream>>>(
        X, (const float*)d_in[62], (const float*)d_in[63], (float*)d_out);
}
